// vPyrochlore_41979010351318
// MI455X (gfx1250) — matrix operand order repaired and checked by reading, not compiled, not run on hardware
//
#include <hip/hip_runtime.h>
#include <hip/hip_bf16.h>

#define LDIM 96
#define NSITES (LDIM*LDIM*LDIM)

typedef __attribute__((ext_vector_type(16))) _Float16 v16h;
typedef __attribute__((ext_vector_type(8)))  _Float16 v8h;
typedef __attribute__((ext_vector_type(8)))  float    v8f;

// workspace layout (bytes)
#define WS_ACC    0                         // 32 f32 partial channel sums
#define WS_WT1    256                       // conv1 B tiles: 14 chunks * 1024 B
#define WS_WT2    16384                     // conv2 B tiles: 14 chunks * 2048 B
#define WS_BONDS  65536                     // NSITES * 16 f16 (ch 12..15 zero)
#define WS_Y1     (WS_BONDS + NSITES*32)    // NSITES * 16 f16

__device__ __forceinline__ int wrapc(int v) {
    return v < 0 ? v + LDIM : (v >= LDIM ? v - LDIM : v);
}

__device__ __forceinline__ float elu1(float v) {
    return v > 0.f ? v : (__expf(v) - 1.f);
}

// ---------------------------------------------------------------------------
// Weight prep: transpose w1/w2 into per-lane WMMA B-tile layout, f32 -> f16.
// wt1 half-index: c*512 + khalf*256 + n*16 + ch      (14 chunks)
// wt2 half-index: c*1024 + nt*512 + khalf*256 + n*16 + ch
// slot (khalf, ch): tap 2c + (ch>>3), channel khalf*8 + (ch&7) (table K order).
// ---------------------------------------------------------------------------
__global__ void prep_weights(const float* __restrict__ w1, const float* __restrict__ w2,
                             _Float16* __restrict__ wt1, _Float16* __restrict__ wt2) {
    int tid = blockIdx.x * blockDim.x + threadIdx.x;
    if (tid < 14 * 2 * 16 * 16) {              // 7168 conv1 tile elements
        int ch = tid & 15, n = (tid >> 4) & 15, kh = (tid >> 8) & 1, c = tid >> 9;
        int tap = 2 * c + (ch >> 3), cc = kh * 8 + (ch & 7);
        float v = 0.f;
        if (tap <= 26 && cc < 12) v = w1[tap * 192 + cc * 16 + n]; // (27,12,16)
        wt1[tid] = (_Float16)v;
    }
    int t2 = tid - 7168;
    if (t2 >= 0 && t2 < 14 * 2 * 2 * 16 * 16) { // 14336 conv2 tile elements
        int ch = t2 & 15, n = (t2 >> 4) & 15, kh = (t2 >> 8) & 1;
        int nt = (t2 >> 9) & 1, c = t2 >> 10;
        int tap = 2 * c + (ch >> 3), cc = kh * 8 + (ch & 7);
        float v = 0.f;
        if (tap <= 26) v = w2[tap * 512 + cc * 32 + nt * 16 + n]; // (27,16,32)
        wt2[t2] = (_Float16)v;
    }
}

// ---------------------------------------------------------------------------
// Bonds: per site, 12 dot products of 2-vectors, padded to 16 f16 channels.
// ---------------------------------------------------------------------------
__global__ void bonds_kernel(const int* __restrict__ lx, const float* __restrict__ t2,
                             _Float16* __restrict__ bonds) {
    int site = blockIdx.x * 256 + threadIdx.x;
    int x = site % LDIM;
    int y = (site / LDIM) % LDIM;
    int z = site / (LDIM * LDIM);

    float t00 = t2[0], t01 = t2[1], t10 = t2[2], t11 = t2[3];

    int s4 = site * 4;
    int c0 = lx[s4 + 0], c1 = lx[s4 + 1], c2 = lx[s4 + 2], c3 = lx[s4 + 3];
    int zm = z == 0 ? LDIM - 1 : z - 1;
    int ym = y == 0 ? LDIM - 1 : y - 1;
    int xm = x == 0 ? LDIM - 1 : x - 1;
    int cr1 = lx[(((zm * LDIM) + y) * LDIM + x) * 4 + 1]; // roll v1 along z
    int cr2 = lx[(((z * LDIM) + ym) * LDIM + x) * 4 + 2]; // roll v2 along y
    int cr3 = lx[(((z * LDIM) + y) * LDIM + xm) * 4 + 3]; // roll v3 along x

    float v0x = c0 ? t10 : t00, v0y = c0 ? t11 : t01;
    float v1x = c1 ? t10 : t00, v1y = c1 ? t11 : t01;
    float v2x = c2 ? t10 : t00, v2y = c2 ? t11 : t01;
    float v3x = c3 ? t10 : t00, v3y = c3 ? t11 : t01;
    float r1x = cr1 ? t10 : t00, r1y = cr1 ? t11 : t01;
    float r2x = cr2 ? t10 : t00, r2y = cr2 ? t11 : t01;
    float r3x = cr3 ? t10 : t00, r3y = cr3 ? t11 : t01;

    float d[12];
    d[0]  = v0x * v1x + v0y * v1y;
    d[1]  = v0x * v2x + v0y * v2y;
    d[2]  = v0x * v3x + v0y * v3y;
    d[3]  = v0x * r1x + v0y * r1y;
    d[4]  = v0x * r2x + v0y * r2y;
    d[5]  = v0x * r3x + v0y * r3y;
    d[6]  = v1x * v2x + v1y * v2y;
    d[7]  = v2x * v3x + v2y * v3y;
    d[8]  = v3x * v1x + v3y * v1y;
    d[9]  = r1x * r2x + r1y * r2y;
    d[10] = r2x * r3x + r2y * r3y;
    d[11] = r3x * r1x + r3y * r1y;

    v8h lo, hi;
#pragma unroll
    for (int i = 0; i < 8; ++i) lo[i] = (_Float16)d[i];
#pragma unroll
    for (int i = 0; i < 4; ++i) hi[i] = (_Float16)d[8 + i];
#pragma unroll
    for (int i = 4; i < 8; ++i) hi[i] = (_Float16)0.f;

    *(v8h*)(bonds + site * 16)     = lo;
    *(v8h*)(bonds + site * 16 + 8) = hi;
}

// ---------------------------------------------------------------------------
// conv1: implicit GEMM, M=16 x-positions, N=16 out channels, 14 K=32 chunks.
// One wave per output tile; 8 waves per block. Wrap/scale math hoisted; all
// offsets unsigned so loads can use the saddr + u32-voffset addressing form.
// ---------------------------------------------------------------------------
__global__ void conv1_kernel(const _Float16* __restrict__ bonds,
                             const _Float16* __restrict__ wt1,
                             const float* __restrict__ b1,
                             _Float16* __restrict__ y1) {
    int lane = threadIdx.x & 31;
    int tile = blockIdx.x * 8 + (threadIdx.x >> 5);
    int seg = tile % 6;
    int y = (tile / 6) % LDIM;
    int z = tile / (6 * LDIM);
    int x0 = seg * 16;
    int m = lane & 15;
    int khalf = lane >> 4;

    // wrapped offsets in f16-element units (site stride = 16 halves);
    // khalf K-half bias folded into X[]; all non-negative -> unsigned
    unsigned Z[3], Y[3], X[3];
    Z[0] = (unsigned)wrapc(z - 1) * (LDIM * LDIM * 16u);
    Z[1] = (unsigned)z * (LDIM * LDIM * 16u);
    Z[2] = (unsigned)wrapc(z + 1) * (LDIM * LDIM * 16u);
    Y[0] = (unsigned)wrapc(y - 1) * (LDIM * 16u);
    Y[1] = (unsigned)y * (LDIM * 16u);
    Y[2] = (unsigned)wrapc(y + 1) * (LDIM * 16u);
    int xm = x0 + m;
    X[0] = (unsigned)wrapc(xm - 1) * 16u + khalf * 8u;
    X[1] = (unsigned)xm * 16u + khalf * 8u;
    X[2] = (unsigned)wrapc(xm + 1) * 16u + khalf * 8u;

    const _Float16* wb = wt1 + khalf * 256 + m * 16;   // n == lane&15 == m

    v8f acc = {};
#pragma unroll
    for (int c = 0; c < 14; ++c) {
        const int ta = 2 * c;
        const int tb = (2 * c + 1 <= 26) ? 2 * c + 1 : 26; // dup tap, zero weights
        unsigned oa = Z[ta / 9] + Y[(ta / 3) % 3] + X[ta % 3];
        unsigned ob = Z[tb / 9] + Y[(tb / 3) % 3] + X[tb % 3];

        v8h alo = *(const v8h*)(bonds + oa);
        v8h ahi = *(const v8h*)(bonds + ob);
        v16h a = __builtin_shufflevector(alo, ahi, 0,1,2,3,4,5,6,7,8,9,10,11,12,13,14,15);
        v16h b = *(const v16h*)(wb + c * 512);

        acc = __builtin_amdgcn_wmma_f32_16x16x32_f16(false, a, false, b,
                                                     (short)0, acc, false, false);
    }

    float bias = b1[m];
    // out half-index: (z*L*L + y*L + x0 + mm)*16 + n,  mm = j + 8*khalf, n = m
    _Float16* outp = y1 + (Z[1] + Y[1] + (unsigned)x0 * 16u + khalf * 128u + (unsigned)m);
#pragma unroll
    for (int j = 0; j < 8; ++j) {
        float v = elu1(acc[j] + bias);
        outp[j * 16] = (_Float16)v;
    }
}

// ---------------------------------------------------------------------------
// conv2: same structure, N=32 (two accumulators), then ELU + channel-sum
// reduction into 32 global partial sums.
// ---------------------------------------------------------------------------
__global__ void conv2_kernel(const _Float16* __restrict__ y1,
                             const _Float16* __restrict__ wt2,
                             const float* __restrict__ b2,
                             float* __restrict__ acc_out) {
    __shared__ float sacc[32];
    if (threadIdx.x < 32) sacc[threadIdx.x] = 0.f;
    __syncthreads();

    int lane = threadIdx.x & 31;
    int tile = blockIdx.x * 8 + (threadIdx.x >> 5);
    int seg = tile % 6;
    int y = (tile / 6) % LDIM;
    int z = tile / (6 * LDIM);
    int x0 = seg * 16;
    int m = lane & 15;
    int khalf = lane >> 4;

    unsigned Z[3], Y[3], X[3];
    Z[0] = (unsigned)wrapc(z - 1) * (LDIM * LDIM * 16u);
    Z[1] = (unsigned)z * (LDIM * LDIM * 16u);
    Z[2] = (unsigned)wrapc(z + 1) * (LDIM * LDIM * 16u);
    Y[0] = (unsigned)wrapc(y - 1) * (LDIM * 16u);
    Y[1] = (unsigned)y * (LDIM * 16u);
    Y[2] = (unsigned)wrapc(y + 1) * (LDIM * 16u);
    int xm = x0 + m;
    X[0] = (unsigned)wrapc(xm - 1) * 16u + khalf * 8u;
    X[1] = (unsigned)xm * 16u + khalf * 8u;
    X[2] = (unsigned)wrapc(xm + 1) * 16u + khalf * 8u;

    const _Float16* wb = wt2 + khalf * 256 + m * 16;

    v8f acc0 = {}, acc1 = {};
#pragma unroll
    for (int c = 0; c < 14; ++c) {
        const int ta = 2 * c;
        const int tb = (2 * c + 1 <= 26) ? 2 * c + 1 : 26;
        unsigned oa = Z[ta / 9] + Y[(ta / 3) % 3] + X[ta % 3];
        unsigned ob = Z[tb / 9] + Y[(tb / 3) % 3] + X[tb % 3];

        v8h alo = *(const v8h*)(y1 + oa);
        v8h ahi = *(const v8h*)(y1 + ob);
        v16h a = __builtin_shufflevector(alo, ahi, 0,1,2,3,4,5,6,7,8,9,10,11,12,13,14,15);
        v16h b0 = *(const v16h*)(wb + c * 1024);
        v16h b1 = *(const v16h*)(wb + c * 1024 + 512);

        acc0 = __builtin_amdgcn_wmma_f32_16x16x32_f16(false, a, false, b0,
                                                      (short)0, acc0, false, false);
        acc1 = __builtin_amdgcn_wmma_f32_16x16x32_f16(false, a, false, b1,
                                                      (short)0, acc1, false, false);
    }

    float bias0 = b2[m], bias1 = b2[16 + m];
    float s0 = 0.f, s1 = 0.f;
#pragma unroll
    for (int j = 0; j < 8; ++j) {
        s0 += elu1(acc0[j] + bias0);
        s1 += elu1(acc1[j] + bias1);
    }
    // combine the two M-halves (lane l and l^16 share channel n = lane&15)
    s0 += __shfl_xor(s0, 16, 32);
    s1 += __shfl_xor(s1, 16, 32);
    if (lane < 16) {
        atomicAdd(&sacc[m], s0);
        atomicAdd(&sacc[16 + m], s1);
    }
    __syncthreads();
    if (threadIdx.x < 32) atomicAdd(&acc_out[threadIdx.x], sacc[threadIdx.x]);
}

// ---------------------------------------------------------------------------
// Final: out = sum_c (acc[c]/L^3) * wd[c]
// ---------------------------------------------------------------------------
__global__ void finalize_kernel(const float* __restrict__ acc,
                                const float* __restrict__ wd,
                                float* __restrict__ out) {
    if (threadIdx.x == 0 && blockIdx.x == 0) {
        float s = 0.f;
        const float inv = 1.f / (float)NSITES;
        for (int c = 0; c < 32; ++c) s += acc[c] * inv * wd[c];
        out[0] = s;
    }
}

extern "C" void kernel_launch(void* const* d_in, const int* in_sizes, int n_in,
                              void* d_out, int out_size, void* d_ws, size_t ws_size,
                              hipStream_t stream) {
    const int*   lx = (const int*)d_in[0];
    const float* t2 = (const float*)d_in[1];
    const float* w1 = (const float*)d_in[2];
    const float* b1 = (const float*)d_in[3];
    const float* w2 = (const float*)d_in[4];
    const float* b2 = (const float*)d_in[5];
    const float* wd = (const float*)d_in[6];

    char* ws = (char*)d_ws;
    float*    acc   = (float*)(ws + WS_ACC);
    _Float16* wt1   = (_Float16*)(ws + WS_WT1);
    _Float16* wt2   = (_Float16*)(ws + WS_WT2);
    _Float16* bonds = (_Float16*)(ws + WS_BONDS);
    _Float16* y1    = (_Float16*)(ws + WS_Y1);

    hipMemsetAsync(acc, 0, 32 * sizeof(float), stream);
    prep_weights<<<84, 256, 0, stream>>>(w1, w2, wt1, wt2);
    bonds_kernel<<<NSITES / 256, 256, 0, stream>>>(lx, t2, bonds);
    conv1_kernel<<<NSITES / 16 / 8, 256, 0, stream>>>(bonds, wt1, b1, y1);
    conv2_kernel<<<NSITES / 16 / 8, 256, 0, stream>>>(y1, wt2, b2, acc);
    finalize_kernel<<<1, 1, 0, stream>>>(acc, wd, (float*)d_out);
}
